// MultiHeadAttention_1314259992869
// MI455X (gfx1250) — matrix operand order repaired and checked by reading, not compiled, not run on hardware
//
#include <hip/hip_runtime.h>
#include <hip/hip_bf16.h>

typedef _Float16 half_t;
typedef __attribute__((ext_vector_type(16))) _Float16 v16h;
typedef __attribute__((ext_vector_type(8)))  _Float16 v8h;
typedef __attribute__((ext_vector_type(4)))  _Float16 v4h;
typedef __attribute__((ext_vector_type(8)))  float    v8f;
typedef __attribute__((ext_vector_type(4)))  float    v4f;
typedef __attribute__((ext_vector_type(4)))  unsigned int u32x4;
typedef __attribute__((ext_vector_type(8)))  int          i32x8;
typedef __attribute__((ext_vector_type(4)))  int          i32x4;

#define B_  4
#define N_  2048
#define E_  1024
#define H_  16
#define D_  64
#define M_  (B_*N_)   /* 8192 token rows */

// ---------------------------------------------------------------------------
// Fragment loaders using this file's contiguous K order (same for A and B; not the ISA table order).
// A 16x32 f16 (MxK): lane = row(0..15) | hi; elems e -> K = hi*16+e.
//                    Two b128 loads.
// B 32x16 f16 (KxN): lane = col(0..15) | hi; elems e -> K = hi*16+e.
//                    Contiguous when tile stored [col][k]. Two b128 loads.
// C/D 16x16 f32:     elem i -> row = hi*8+i, col = lane&15.
// ---------------------------------------------------------------------------
__device__ __forceinline__ v16h frag_a16x32(const half_t* p, int ld) {
  const int lane = threadIdx.x & 31;
  const int row  = lane & 15;
  const int hi   = lane >> 4;
  const half_t* rp = p + row * ld + hi * 16;
  v8h lo = *(const v8h*)rp;          // K = hi*16 .. +7
  v8h hh = *(const v8h*)(rp + 8);    // K = hi*16+8 .. +15
  v16h r;
#pragma unroll
  for (int i = 0; i < 8; ++i) { r[i] = lo[i]; r[8 + i] = hh[i]; }
  return r;
}

__device__ __forceinline__ v16h frag_b32x16(const half_t* p, int ld) {
  // p points at tile stored column-major-as-rows: p[col*ld + k]
  const int lane = threadIdx.x & 31;
  const int col  = lane & 15;
  const int hi   = lane >> 4;
  const half_t* cp = p + col * ld + hi * 16;
  v8h lo = *(const v8h*)cp;          // K = hi*16 .. +7
  v8h hh = *(const v8h*)(cp + 8);    // K = hi*16+8 .. +15
  v16h r;
#pragma unroll
  for (int i = 0; i < 8; ++i) { r[i] = lo[i]; r[8 + i] = hh[i]; }
  return r;
}

__device__ __forceinline__ v8f wmma_f16(v16h a, v16h b, v8f c) {
  return __builtin_amdgcn_wmma_f32_16x16x32_f16(
      /*neg_a=*/false, a, /*neg_b=*/false, b,
      /*c_mod=*/(short)0, c, /*reuse_a=*/false, /*reuse_b=*/false);
}

// ---------------------------------------------------------------------------
// DPP16 butterfly reductions within each 16-lane half (pure VALU, no LDS).
// 0xB1 = quad_perm [1,0,3,2] (xor1), 0x4E = quad_perm [2,3,0,1] (xor2),
// 0x141 = row_half_mirror (pairs 8-groups), 0x140 = row_mirror (pairs 16).
// ---------------------------------------------------------------------------
template <int CTRL>
__device__ __forceinline__ float dpp_mov(float v) {
  return __int_as_float(__builtin_amdgcn_update_dpp(
      __float_as_int(v), __float_as_int(v), CTRL, 0xF, 0xF, true));
}
__device__ __forceinline__ float red_max16(float v) {
  v = fmaxf(v, dpp_mov<0xB1>(v));
  v = fmaxf(v, dpp_mov<0x4E>(v));
  v = fmaxf(v, dpp_mov<0x141>(v));
  v = fmaxf(v, dpp_mov<0x140>(v));
  return v;
}
__device__ __forceinline__ float red_sum16(float v) {
  v += dpp_mov<0xB1>(v);
  v += dpp_mov<0x4E>(v);
  v += dpp_mov<0x141>(v);
  v += dpp_mov<0x140>(v);
  return v;
}

// ---------------------------------------------------------------------------
// TDM: async-load a 64x64 f16 tile (row stride 64 elems) from global to LDS,
// padding +16B after every 128B row -> LDS row stride 144B ([64][72] layout).
// Descriptor per CDNA5 ISA:
//  g0: [1:0]count=1 | [63:32]lds_addr | [120:64]global_addr | [127:126]type=2
//  g1: [17:16]data_size=1(2B) | [20]pad_en | [24:22]pad_interval=4 (32 dwords)
//      | [31:25]pad_amount=3 (4 dwords) | [79:48]tensor_dim0=64
//      | [111:80]tensor_dim1=2048 | [127:112]tile_dim0=64 | [143:128]tile_dim1=64
//      | [207:160]tensor_dim0_stride=64
//  remaining groups zero (2-D tensor). Tracked by TENSORcnt.
// ---------------------------------------------------------------------------
__device__ __forceinline__ void tdm_load_tile_64x64(const half_t* gsrc,
                                                    unsigned lds_off) {
  unsigned long long ga = (unsigned long long)(size_t)gsrc;
  u32x4 g0;
  g0[0] = 1u;                                               // count=1 (valid)
  g0[1] = lds_off;                                          // lds_addr
  g0[2] = (unsigned)ga;                                     // global_addr lo
  g0[3] = (unsigned)((ga >> 32) & 0x1FFFFFFull) | (2u << 30); // hi | type=2
  i32x8 g1;
  g1[0] = (1 << 16) | (1 << 20) | (4 << 22) | (3 << 25);
  g1[1] = (int)(64u << 16);          // tensor_dim0 = 64   (bits 79:48)
  g1[2] = (int)((unsigned)N_ << 16); // tensor_dim1 = 2048 (bits 111:80)
  g1[3] = (int)(64u << 16);          // tile_dim0 = 64     (bits 127:112)
  g1[4] = 64;                        // tile_dim1 = 64     (bits 143:128)
  g1[5] = 64;                        // tensor_dim0_stride (bits 207:160)
  g1[6] = 0;
  g1[7] = 0;
  i32x4 z4 = {0, 0, 0, 0};
  i32x8 z8 = {0, 0, 0, 0, 0, 0, 0, 0};
  // amdgpu-toolchain (clang-23) 6-arg form: (g0, g1, g2, g3, g4, cpol)
  __builtin_amdgcn_tensor_load_to_lds(g0, g1, z4, z4, z8, 0);
}

// ---------------------------------------------------------------------------
// Fused QKV projection: one A-tile load feeds Wq/Wk/Wv (12 WMMA / wave / step).
// Block = 256 threads (8 waves), tile 128(M) x 64(N), K-step 32.
// Outputs f16 in [b,h,n,d] head-split layout; Q pre-scaled by 1/sqrt(E).
// ---------------------------------------------------------------------------
__global__ __launch_bounds__(256)
void qkv_kernel(const float* __restrict__ x,
                const float* __restrict__ Wq, const float* __restrict__ bq,
                const float* __restrict__ Wk, const float* __restrict__ bk,
                const float* __restrict__ Wv, const float* __restrict__ bv,
                half_t* __restrict__ Qw, half_t* __restrict__ Kw,
                half_t* __restrict__ Vw, float qscale)
{
  __shared__ alignas(16) half_t As[128][40];     // [row][k]
  __shared__ alignas(16) half_t Ws[3][64][40];   // transposed: [j][n][k]

  const int tid  = threadIdx.x;
  const int lane = tid & 31;
  const int w    = tid >> 5;
  const int m0   = blockIdx.x * 128;
  const int n0   = blockIdx.y * 64;
  const int wr   = (w >> 1) * 32;
  const int wc   = (w & 1) * 32;

  v8f acc[3][2][2] = {};

  for (int kk = 0; kk < E_; kk += 32) {
    // A tile: 128x32 f32 -> f16
#pragma unroll
    for (int it = 0; it < 4; ++it) {
      int idx = tid + it * 256;
      int r   = idx >> 3;
      int c   = (idx & 7) * 4;
      v4f v = *(const v4f*)(x + (size_t)(m0 + r) * E_ + kk + c);
      As[r][c + 0] = (half_t)v[0]; As[r][c + 1] = (half_t)v[1];
      As[r][c + 2] = (half_t)v[2]; As[r][c + 3] = (half_t)v[3];
    }
    // three W tiles: 32x64 each, stored transposed [n][k]
#pragma unroll
    for (int j = 0; j < 3; ++j) {
      const float* Wj = (j == 0) ? Wq : ((j == 1) ? Wk : Wv);
#pragma unroll
      for (int it = 0; it < 2; ++it) {
        int idx = tid + it * 256;
        int k   = idx >> 4;
        int n   = (idx & 15) * 4;
        v4f v = *(const v4f*)(Wj + (size_t)(kk + k) * E_ + n0 + n);
        Ws[j][n + 0][k] = (half_t)v[0]; Ws[j][n + 1][k] = (half_t)v[1];
        Ws[j][n + 2][k] = (half_t)v[2]; Ws[j][n + 3][k] = (half_t)v[3];
      }
    }
    __syncthreads();

    v16h a0 = frag_a16x32(&As[wr][0], 40);
    v16h a1 = frag_a16x32(&As[wr + 16][0], 40);
#pragma unroll
    for (int j = 0; j < 3; ++j) {
      v16h b0 = frag_b32x16(&Ws[j][wc][0], 40);
      v16h b1 = frag_b32x16(&Ws[j][wc + 16][0], 40);
      acc[j][0][0] = wmma_f16(a0, b0, acc[j][0][0]);
      acc[j][0][1] = wmma_f16(a0, b1, acc[j][0][1]);
      acc[j][1][0] = wmma_f16(a1, b0, acc[j][1][0]);
      acc[j][1][1] = wmma_f16(a1, b1, acc[j][1][1]);
    }
    __syncthreads();
  }

  // epilogue: head-split f16 stores
  const int hi  = lane >> 4;
  const int cth = lane & 15;
#pragma unroll
  for (int j = 0; j < 3; ++j) {
    half_t* out = (j == 0) ? Qw : ((j == 1) ? Kw : Vw);
    const float* bias = (j == 0) ? bq : ((j == 1) ? bk : bv);
    const float sc = (j == 0) ? qscale : 1.0f;
#pragma unroll
    for (int ji = 0; ji < 2; ++ji) {
#pragma unroll
      for (int jb = 0; jb < 2; ++jb) {
#pragma unroll
        for (int i = 0; i < 8; ++i) {
          int   m   = m0 + wr + ji * 16 + hi * 8 + i;
          int   col = n0 + wc + jb * 16 + cth;
          float v   = (acc[j][ji][jb][i] + bias[col]) * sc;
          int b = m >> 11, n = m & (N_ - 1);
          int h = col >> 6, d = col & (D_ - 1);
          out[((size_t)(b * H_ + h) * N_ + n) * D_ + d] = (half_t)v;
        }
      }
    }
  }
}

// ---------------------------------------------------------------------------
// Flash attention: grid (b*h = 64, n/128 = 16), block 256 = 8 waves.
// Wave owns 16 query rows x d=64. Per 64-key block:
//   S(16x64)  = Q * K^T : 8 WMMAs  (K tile TDM-prefetched, double-buffered)
//   online softmax (DPP16 butterflies)
//   O(16x64) += P * V   : 8 WMMAs  (V tile transposed [d][key] in LDS)
// Q pre-scaled by 1/sqrt(E). Output f16 [b,n,e].
// ---------------------------------------------------------------------------
__global__ __launch_bounds__(256)
void attn_kernel(const half_t* __restrict__ Q, const half_t* __restrict__ Km,
                 const half_t* __restrict__ V, half_t* __restrict__ O)
{
  __shared__ alignas(16) half_t Kt[2][64][72];   // [buf][key][d], TDM-filled
  __shared__ alignas(16) half_t Vt[64][72];      // [d][key] (transposed)
  __shared__ alignas(16) half_t Pb[8][16][72];   // per-wave P scratch

  const int tid  = threadIdx.x;
  const int lane = tid & 31;
  const int w    = tid >> 5;
  const int bh   = blockIdx.x;                   // b*16 + h
  const int q0   = blockIdx.y * 128 + w * 16;
  const size_t base = (size_t)bh * N_ * D_;

  const int hi  = lane >> 4;
  const int cth = lane & 15;

  // Q fragments stay in registers for the whole kernel (rows q0..q0+15)
  v16h qf0 = frag_a16x32(Q + base + (size_t)q0 * D_, D_);       // d 0..31
  v16h qf1 = frag_a16x32(Q + base + (size_t)q0 * D_ + 32, D_);  // d 32..63

  // prime the TDM pipeline: wave 0 DMAs key block 0 into Kt[0]
  if (w == 0)
    tdm_load_tile_64x64(Km + base, (unsigned)(size_t)&Kt[0][0][0]);

  v8f   o[4] = {};
  float mrow[8], lrow[8];
#pragma unroll
  for (int i = 0; i < 8; ++i) { mrow[i] = -1e30f; lrow[i] = 0.f; }

  int cur = 0;
  for (int kb = 0; kb < N_; kb += 64) {
    { // cooperative V tile load (transposed): 64 keys x 64 d
      int key = tid >> 2;
      int d8  = (tid & 3) * 16;
      const half_t* vp = V + base + (size_t)(kb + key) * D_ + d8;
      v8h vv0 = *(const v8h*)vp;
      v8h vv1 = *(const v8h*)(vp + 8);
#pragma unroll
      for (int i = 0; i < 8; ++i) {
        Vt[d8 + i][key]     = vv0[i];
        Vt[d8 + 8 + i][key] = vv1[i];
      }
    }
    // wave 0: prefetch next K block, then drain until current block landed.
    // TENSORcnt completes in-order -> wait(<=1) guarantees block kb is ready.
    if (w == 0) {
      if (kb + 64 < N_) {
        tdm_load_tile_64x64(Km + base + (size_t)(kb + 64) * D_,
                            (unsigned)(size_t)&Kt[cur ^ 1][0][0]);
        __builtin_amdgcn_s_wait_tensorcnt(1);
      } else {
        __builtin_amdgcn_s_wait_tensorcnt(0);
      }
    }
    __syncthreads();

    // S = Q * K^T   (B-fragment column = key position, K-dim = d)
    const half_t* kt = &Kt[cur][0][0];
    v8f s0 = {}, s1 = {}, s2 = {}, s3 = {};
    s0 = wmma_f16(qf0, frag_b32x16(kt, 72), s0);
    s0 = wmma_f16(qf1, frag_b32x16(kt + 32, 72), s0);
    s1 = wmma_f16(qf0, frag_b32x16(kt + 16 * 72, 72), s1);
    s1 = wmma_f16(qf1, frag_b32x16(kt + 16 * 72 + 32, 72), s1);
    s2 = wmma_f16(qf0, frag_b32x16(kt + 32 * 72, 72), s2);
    s2 = wmma_f16(qf1, frag_b32x16(kt + 32 * 72 + 32, 72), s2);
    s3 = wmma_f16(qf0, frag_b32x16(kt + 48 * 72, 72), s3);
    s3 = wmma_f16(qf1, frag_b32x16(kt + 48 * 72 + 32, 72), s3);

    // online softmax: elem i <-> query row hi*8+i; reduce over 16-lane half
#pragma unroll
    for (int i = 0; i < 8; ++i) {
      float mb = red_max16(fmaxf(fmaxf(s0[i], s1[i]), fmaxf(s2[i], s3[i])));
      float mnew  = fmaxf(mrow[i], mb);
      float alpha = __expf(mrow[i] - mnew);
      float p0 = __expf(s0[i] - mnew);
      float p1 = __expf(s1[i] - mnew);
      float p2 = __expf(s2[i] - mnew);
      float p3 = __expf(s3[i] - mnew);
      float ps = red_sum16((p0 + p1) + (p2 + p3));
      lrow[i] = lrow[i] * alpha + ps;
      mrow[i] = mnew;
#pragma unroll
      for (int j = 0; j < 4; ++j) o[j][i] *= alpha;
      int r = hi * 8 + i;
      Pb[w][r][cth]      = (half_t)p0;   // keys  0..15
      Pb[w][r][16 + cth] = (half_t)p1;   // keys 16..31
      Pb[w][r][32 + cth] = (half_t)p2;   // keys 32..47
      Pb[w][r][48 + cth] = (half_t)p3;   // keys 48..63
    }
    // per-wave LDS round-trip: make the 16x64 P tile visible before reload
    asm volatile("s_wait_dscnt 0" ::: "memory");

    v16h pf0 = frag_a16x32(&Pb[w][0][0], 72);   // A = P, keys  0..31
    v16h pf1 = frag_a16x32(&Pb[w][0][32], 72);  // A = P, keys 32..63
#pragma unroll
    for (int j = 0; j < 4; ++j) {
      o[j] = wmma_f16(pf0, frag_b32x16(&Vt[j * 16][0], 72),  o[j]);
      o[j] = wmma_f16(pf1, frag_b32x16(&Vt[j * 16][32], 72), o[j]);
    }
    __syncthreads();   // protect Vt (and retire Kt[cur]) before next block
    cur ^= 1;
  }

  // normalize and write O in [b, n, e] f16 layout
  const int b = bh >> 4;
  const int h = bh & 15;
#pragma unroll
  for (int i = 0; i < 8; ++i) {
    float inv = 1.0f / lrow[i];
    int   n   = q0 + hi * 8 + i;
    half_t* dst = O + ((size_t)b * N_ + n) * E_ + h * D_;
#pragma unroll
    for (int j = 0; j < 4; ++j)
      dst[j * 16 + cth] = (half_t)(o[j][i] * inv);
  }
}

// ---------------------------------------------------------------------------
// Output projection: out[M_ x E_] f32 = O(f16) * Wo + bo.
// Block = 256 threads, tile 128(M) x 64(N), K-step 32.
// ---------------------------------------------------------------------------
__global__ __launch_bounds__(256)
void oproj_kernel(const half_t* __restrict__ A, const float* __restrict__ W,
                  const float* __restrict__ bias, float* __restrict__ out)
{
  __shared__ alignas(16) half_t As[128][40];
  __shared__ alignas(16) half_t Ws[64][40];

  const int tid  = threadIdx.x;
  const int lane = tid & 31;
  const int w    = tid >> 5;
  const int m0   = blockIdx.x * 128;
  const int n0   = blockIdx.y * 64;
  const int wr   = (w >> 1) * 32;
  const int wc   = (w & 1) * 32;

  v8f acc[2][2] = {};

  for (int kk = 0; kk < E_; kk += 32) {
#pragma unroll
    for (int it = 0; it < 4; ++it) {
      int idx = tid + it * 256;
      int r   = idx >> 3;
      int c   = (idx & 7) * 4;
      v4h v = *(const v4h*)(A + (size_t)(m0 + r) * E_ + kk + c);
      As[r][c + 0] = v[0]; As[r][c + 1] = v[1];
      As[r][c + 2] = v[2]; As[r][c + 3] = v[3];
    }
#pragma unroll
    for (int it = 0; it < 2; ++it) {
      int idx = tid + it * 256;
      int k   = idx >> 4;
      int n   = (idx & 15) * 4;
      v4f v = *(const v4f*)(W + (size_t)(kk + k) * E_ + n0 + n);
      Ws[n + 0][k] = (half_t)v[0]; Ws[n + 1][k] = (half_t)v[1];
      Ws[n + 2][k] = (half_t)v[2]; Ws[n + 3][k] = (half_t)v[3];
    }
    __syncthreads();

    v16h a0 = frag_a16x32(&As[wr][0], 40);
    v16h a1 = frag_a16x32(&As[wr + 16][0], 40);
    v16h b0 = frag_b32x16(&Ws[wc][0], 40);
    v16h b1 = frag_b32x16(&Ws[wc + 16][0], 40);
    acc[0][0] = wmma_f16(a0, b0, acc[0][0]);
    acc[0][1] = wmma_f16(a0, b1, acc[0][1]);
    acc[1][0] = wmma_f16(a1, b0, acc[1][0]);
    acc[1][1] = wmma_f16(a1, b1, acc[1][1]);
    __syncthreads();
  }

  const int hi  = lane >> 4;
  const int cth = lane & 15;
#pragma unroll
  for (int ji = 0; ji < 2; ++ji) {
#pragma unroll
    for (int jb = 0; jb < 2; ++jb) {
#pragma unroll
      for (int i = 0; i < 8; ++i) {
        int m   = m0 + wr + ji * 16 + hi * 8 + i;
        int col = n0 + wc + jb * 16 + cth;
        out[(size_t)m * E_ + col] = acc[ji][jb][i] + bias[col];
      }
    }
  }
}

// ---------------------------------------------------------------------------
extern "C" void kernel_launch(void* const* d_in, const int* in_sizes, int n_in,
                              void* d_out, int out_size, void* d_ws, size_t ws_size,
                              hipStream_t stream)
{
  const float* x  = (const float*)d_in[0];
  const float* Wq = (const float*)d_in[1];
  const float* bq = (const float*)d_in[2];
  const float* Wk = (const float*)d_in[3];
  const float* bk = (const float*)d_in[4];
  const float* Wv = (const float*)d_in[5];
  const float* bv = (const float*)d_in[6];
  const float* Wo = (const float*)d_in[7];
  const float* bo = (const float*)d_in[8];

  // workspace: Q,K,V in [b,h,n,d] f16 (16 MB each) + O in [b,n,e] f16 (16 MB)
  half_t* Qw = (half_t*)d_ws;
  half_t* Kw = Qw + (size_t)M_ * E_;
  half_t* Vw = Kw + (size_t)M_ * E_;
  half_t* Ow = Vw + (size_t)M_ * E_;

  dim3 bb(256);
  const float qscale = 0.03125f;  // 1/sqrt(1024), folded into Q

  qkv_kernel<<<dim3(M_ / 128, E_ / 64), bb, 0, stream>>>(
      x, Wq, bq, Wk, bk, Wv, bv, Qw, Kw, Vw, qscale);

  attn_kernel<<<dim3(B_ * H_, N_ / 128), bb, 0, stream>>>(Qw, Kw, Vw, Ow);

  oproj_kernel<<<dim3(M_ / 128, E_ / 64), bb, 0, stream>>>(Ow, Wo, bo, (float*)d_out);
}
